// S4Layer_44392781972052
// MI455X (gfx1250) — matrix operand order repaired and checked by reading, not compiled, not run on hardware
//
#include <hip/hip_runtime.h>
#include <cstdint>

#define D_MODEL 2048
#define D_STATE 128
#define SEQ_T   2048
#define BATCH   8
#define BT      (BATCH * SEQ_T)     // 16384 rows
#define CHUNK   32
#define NCHUNK  (SEQ_T / CHUNK)     // 64
#define LN_EPS  1e-5f
#define YSTRIDE 2052                // 2048 + 4 pad: makes WMMA->LDS stores conflict-free

typedef __attribute__((ext_vector_type(16))) __bf16 v16bf;
typedef __attribute__((ext_vector_type(8)))  __bf16 v8bf;
typedef __attribute__((ext_vector_type(8)))  float  v8f;
typedef __attribute__((ext_vector_type(16))) float  v16f;

union AFrag { v16bf v; v8bf h[2]; unsigned short s[16]; };
union XF    { v16f v; float4 q[4]; float e[16]; };

// Scalar fp32 -> bf16 (RNE) via native fptrunc; backend picks v_cvt_pk_bf16_f32
// class hardware where available instead of a 3-op bit-twiddle sequence.
__device__ __forceinline__ unsigned short f2bf(float f) {
  union { __bf16 b; unsigned short s; } u;
  u.b = (__bf16)f;
  return u.s;
}

// ---------------------------------------------------------------------------
// K0: convert B_mat / C_mat fp32 -> bf16 (1 MB total, L2-resident afterwards)
// ---------------------------------------------------------------------------
__global__ void k_cvt_weights(const float* __restrict__ Bm, const float* __restrict__ Cm,
                              uint16_t* __restrict__ B16, uint16_t* __restrict__ C16) {
  int i = blockIdx.x * blockDim.x + threadIdx.x;
  if (i < D_STATE * D_MODEL) {
    B16[i] = f2bf(Bm[i]);
    C16[i] = f2bf(Cm[i]);
  }
}

// ---------------------------------------------------------------------------
// K1: u[bt,n] = sum_d x[bt,d] * B_mat[n,d]     (16384x128 = x(16384x2048) * B^T)
// Block = 128 threads (4 waves). Block owns a 16-row bt tile; wave w computes
// n-tiles {2w, 2w+1}. K-loop over D in steps of 32 with bf16 WMMA, f32 acc.
// ---------------------------------------------------------------------------
__global__ void k_gemm1(const float* __restrict__ x, const uint16_t* __restrict__ B16,
                        float* __restrict__ u) {
  const int bt0  = blockIdx.x * 16;
  const int lane = threadIdx.x & 31;
  const int wave = threadIdx.x >> 5;      // 0..3
  const int mrow = lane & 15;             // A-matrix row (M), also B column (N)
  const int hsel = lane >> 4;             // half-wave select

  v8f acc0 = {};
  v8f acc1 = {};
  const float* xrow = x + (size_t)(bt0 + mrow) * D_MODEL;

  for (int d0 = 0; d0 < D_MODEL; d0 += 32) {
    // A fragment: 16x32 bf16 tile of x. Lane holds K = hsel*16 + {0..15}.
    const float4* xq = reinterpret_cast<const float4*>(xrow + d0 + hsel * 16);
    XF xf;
    xf.q[0] = xq[0];
    xf.q[1] = xq[1];
    xf.q[2] = xq[2];
    xf.q[3] = xq[3];
    AFrag a;
    a.v = __builtin_convertvector(xf.v, v16bf);   // vector fptrunc, RNE

    // B fragments: column n = nt*16 + mrow, K-run = hsel*16 + [0,16) -> 16
    // consecutive d's in row n of B_mat (row-major [N,D]): one 32B load.
    const uint16_t* bp0 = B16 + (size_t)((wave * 2 + 0) * 16 + mrow) * D_MODEL + d0 + hsel * 16;
    const uint16_t* bp1 = B16 + (size_t)((wave * 2 + 1) * 16 + mrow) * D_MODEL + d0 + hsel * 16;
    v16bf b0 = *reinterpret_cast<const v16bf*>(bp0);
    v16bf b1 = *reinterpret_cast<const v16bf*>(bp1);

    acc0 = __builtin_amdgcn_wmma_f32_16x16x32_bf16(false, a.v, false, b0, (short)0, acc0, false, false);
    acc1 = __builtin_amdgcn_wmma_f32_16x16x32_bf16(false, a.v, false, b1, (short)0, acc1, false, false);
  }

  // C/D layout: VGPR j -> row (j + 8*hsel), col = lane&15
  #pragma unroll
  for (int j = 0; j < 8; ++j) {
    const size_t r = (size_t)(bt0 + j + hsel * 8) * D_STATE;
    u[r + (wave * 2 + 0) * 16 + mrow] = acc0[j];
    u[r + (wave * 2 + 1) * 16 + mrow] = acc1[j];
  }
}

// ---------------------------------------------------------------------------
// K2: per-chunk local scan (h0 = 0) -> chunk-final carry. 512 blocks x 128.
// All 32 loads are independent and hoisted before the FMA chain.
// ---------------------------------------------------------------------------
__global__ void k_chunk_final(const float* __restrict__ u, const float* __restrict__ logA,
                              float* __restrict__ carry_raw) {
  const int b = blockIdx.x / NCHUNK;
  const int c = blockIdx.x % NCHUNK;
  const int n = threadIdx.x;
  const float a = expf(-expf(logA[n]));
  const float* up = u + (size_t)(b * SEQ_T + c * CHUNK) * D_STATE + n;
  float buf[CHUNK];
  #pragma unroll
  for (int t = 0; t < CHUNK; ++t) buf[t] = up[(size_t)t * D_STATE];
  float h = 0.f;
  #pragma unroll
  for (int t = 0; t < CHUNK; ++t) h = fmaf(h, a, buf[t]);
  carry_raw[(size_t)(b * NCHUNK + c) * D_STATE + n] = h;
}

// ---------------------------------------------------------------------------
// K3: exclusive prefix over the 64 chunk carries with decay A^CHUNK.
// 8 blocks x 128 lanes; loads hoisted (single memory round trip).
// ---------------------------------------------------------------------------
__global__ void k_prefix(const float* __restrict__ carry_raw, const float* __restrict__ logA,
                         float* __restrict__ carry_in) {
  const int b = blockIdx.x;
  const int n = threadIdx.x;
  const float aL = expf(-expf(logA[n]) * (float)CHUNK);   // A_disc^CHUNK
  float buf[NCHUNK];
  #pragma unroll
  for (int c = 0; c < NCHUNK; ++c)
    buf[c] = carry_raw[(size_t)(b * NCHUNK + c) * D_STATE + n];
  float h = 0.f;
  #pragma unroll
  for (int c = 0; c < NCHUNK; ++c) {
    carry_in[(size_t)(b * NCHUNK + c) * D_STATE + n] = h;
    h = fmaf(h, aL, buf[c]);
  }
}

// ---------------------------------------------------------------------------
// K4: re-run chunks with correct carry-in, emit hs as bf16 for GEMM2.
// ---------------------------------------------------------------------------
__global__ void k_scan_store(const float* __restrict__ u, const float* __restrict__ logA,
                             const float* __restrict__ carry_in, uint16_t* __restrict__ hs16) {
  const int b = blockIdx.x / NCHUNK;
  const int c = blockIdx.x % NCHUNK;
  const int n = threadIdx.x;
  const float a = expf(-expf(logA[n]));
  const size_t base = (size_t)(b * SEQ_T + c * CHUNK) * D_STATE + n;
  float buf[CHUNK];
  #pragma unroll
  for (int t = 0; t < CHUNK; ++t) buf[t] = u[base + (size_t)t * D_STATE];
  float h = carry_in[(size_t)(b * NCHUNK + c) * D_STATE + n];
  #pragma unroll
  for (int t = 0; t < CHUNK; ++t) {
    h = fmaf(h, a, buf[t]);
    hs16[base + (size_t)t * D_STATE] = f2bf(h);
  }
}

// ---------------------------------------------------------------------------
// K5: y = hs * C^T + x*(1+Dvec); out = LayerNorm(y).  Fused via 320KB WGP LDS:
// block (256 thr = 8 waves) computes a full 16x2048 strip into LDS, then LN.
// ---------------------------------------------------------------------------
__global__ void k_gemm2_ln(const uint16_t* __restrict__ hs16, const uint16_t* __restrict__ C16,
                           const float* __restrict__ x, const float* __restrict__ Dvec,
                           const float* __restrict__ lnw, const float* __restrict__ lnb,
                           float* __restrict__ out) {
  extern __shared__ float smem[];
  float* ys  = smem;                       // 16 * YSTRIDE floats (128.25 KB)
  float* dvw = ys + 16 * YSTRIDE;          // 1 + Dvec
  float* wln = dvw + D_MODEL;
  float* bln = wln + D_MODEL;

  const int bt0  = blockIdx.x * 16;
  const int tid  = threadIdx.x;
  const int lane = tid & 31;
  const int wave = tid >> 5;               // 0..7
  const int mrow = lane & 15;
  const int hsel = lane >> 4;

  for (int i = tid; i < D_MODEL; i += 256) {
    dvw[i] = 1.0f + Dvec[i];
    wln[i] = lnw[i];
    bln[i] = lnb[i];
  }

  // A fragments: hs rows [bt0..bt0+16) x 128 states = 4 WMMA K-slices.
  AFrag afr[4];
  const uint16_t* hp = hs16 + (size_t)(bt0 + mrow) * D_STATE;
  #pragma unroll
  for (int ka = 0; ka < 4; ++ka) {
    const int kb = ka * 32 + hsel * 16;
    afr[ka].h[0] = *reinterpret_cast<const v8bf*>(hp + kb);
    afr[ka].h[1] = *reinterpret_cast<const v8bf*>(hp + kb + 8);
  }

  // Wave w handles d-tiles w, w+8, ... (16 of 128). B element (k,col) =
  // C_mat[d_col, k]: 16 consecutive states of row d_col -> one 32B load.
  for (int dt = wave; dt < D_MODEL / 16; dt += 8) {
    const int d0 = dt * 16;
    v8f acc = {};
    const uint16_t* cp = C16 + (size_t)(d0 + mrow) * D_STATE + hsel * 16;
    #pragma unroll
    for (int ka = 0; ka < 4; ++ka) {
      v16bf bfrag = *reinterpret_cast<const v16bf*>(cp + ka * 32);
      acc = __builtin_amdgcn_wmma_f32_16x16x32_bf16(false, afr[ka].v, false, bfrag, (short)0, acc, false, false);
    }
    #pragma unroll
    for (int j = 0; j < 8; ++j)
      ys[(j + hsel * 8) * YSTRIDE + d0 + mrow] = acc[j];
  }

  __syncthreads();

  // LayerNorm: wave w handles rows {2w, 2w+1}; wave32 shuffle reductions.
  #pragma unroll
  for (int rr = 0; rr < 2; ++rr) {
    const int r = wave * 2 + rr;
    const float* xr = x + (size_t)(bt0 + r) * D_MODEL;
    float* yr = ys + r * YSTRIDE;

    float s = 0.f;
    for (int i = lane; i < D_MODEL; i += 32) {
      float v = yr[i] + xr[i] * dvw[i];    // + x*(1+D)  (residual + skip term)
      yr[i] = v;
      s += v;
    }
    #pragma unroll
    for (int m = 16; m >= 1; m >>= 1) s += __shfl_xor(s, m, 32);
    const float mu = s * (1.0f / D_MODEL);

    float q = 0.f;
    for (int i = lane; i < D_MODEL; i += 32) {
      float d = yr[i] - mu;
      q += d * d;
    }
    #pragma unroll
    for (int m = 16; m >= 1; m >>= 1) q += __shfl_xor(q, m, 32);
    const float rs = rsqrtf(q * (1.0f / D_MODEL) + LN_EPS);

    float* orow = out + (size_t)(bt0 + r) * D_MODEL;
    for (int i = lane; i < D_MODEL; i += 32)
      orow[i] = (yr[i] - mu) * rs * wln[i] + bln[i];
  }
}

// ---------------------------------------------------------------------------
extern "C" void kernel_launch(void* const* d_in, const int* in_sizes, int n_in,
                              void* d_out, int out_size, void* d_ws, size_t ws_size,
                              hipStream_t stream) {
  (void)in_sizes; (void)n_in; (void)out_size; (void)ws_size;
  const float* x    = (const float*)d_in[0];
  const float* logA = (const float*)d_in[1];
  const float* Bm   = (const float*)d_in[2];
  const float* Cm   = (const float*)d_in[3];
  const float* Dv   = (const float*)d_in[4];
  const float* lnw  = (const float*)d_in[5];
  const float* lnb  = (const float*)d_in[6];
  float* out = (float*)d_out;

  // Workspace carve-up (naturally 256B aligned), ~14.2 MB total.
  char* p = (char*)d_ws;
  uint16_t* B16       = (uint16_t*)p; p += (size_t)D_STATE * D_MODEL * 2;   // 512 KB
  uint16_t* C16       = (uint16_t*)p; p += (size_t)D_MODEL * D_STATE * 2;   // 512 KB
  float*    u         = (float*)p;    p += (size_t)BT * D_STATE * 4;        // 8 MB
  float*    carry_raw = (float*)p;    p += (size_t)BATCH * NCHUNK * D_STATE * 4;
  float*    carry_in  = (float*)p;    p += (size_t)BATCH * NCHUNK * D_STATE * 4;
  uint16_t* hs16      = (uint16_t*)p; p += (size_t)BT * D_STATE * 2;        // 4 MB

  k_cvt_weights<<<(D_STATE * D_MODEL + 255) / 256, 256, 0, stream>>>(Bm, Cm, B16, C16);
  k_gemm1<<<BT / 16, 128, 0, stream>>>(x, B16, u);
  k_chunk_final<<<BATCH * NCHUNK, D_STATE, 0, stream>>>(u, logA, carry_raw);
  k_prefix<<<BATCH, D_STATE, 0, stream>>>(carry_raw, logA, carry_in);
  k_scan_store<<<BATCH * NCHUNK, D_STATE, 0, stream>>>(u, logA, carry_in, hs16);

  const size_t smem_bytes = (size_t)(16 * YSTRIDE + 3 * D_MODEL) * sizeof(float); // ~152 KB
  k_gemm2_ln<<<BT / 16, 256, smem_bytes, stream>>>(hs16, C16, x, Dv, lnw, lnb, out);
}
